// FeatureGrid_6244882448587
// MI455X (gfx1250) — matrix operand order repaired and checked by reading, not compiled, not run on hardware
//
#include <hip/hip_runtime.h>
#include <math.h>

typedef __attribute__((ext_vector_type(16))) _Float16 v16h;
typedef __attribute__((ext_vector_type(8)))  _Float16 v8h;
typedef __attribute__((ext_vector_type(8)))  float    v8f;

#define HIDDEN   64
#define KPAD     72      // padded LDS row stride in halves (144B, conflict-free 16-lane access)
#define NLAYERS  4
#define BLK      128     // 4 waves * 32 lanes; block handles 128 points

#define SHUF16(lo, hi) __builtin_shufflevector((lo), (hi), \
        0, 1, 2, 3, 4, 5, 6, 7, 8, 9, 10, 11, 12, 13, 14, 15)

// Trilinear sample of one feature grid (F=16, G^3, feature-major).
// x-direction as packed float2 loads; per-pair weights pre-scaled -> 2 FMA/feat.
__device__ __forceinline__ void sample_grid(const float* __restrict__ g, int G,
                                            float cx, float cy, float cz,
                                            float* __restrict__ acc /*16*/) {
  float fx = (cx + 1.0f) * 0.5f * (float)(G - 1);
  float fy = (cy + 1.0f) * 0.5f * (float)(G - 1);
  float fz = (cz + 1.0f) * 0.5f * (float)(G - 1);
  float ffx = floorf(fx), ffy = floorf(fy), ffz = floorf(fz);
  float wx = fx - ffx, wy = fy - ffy, wz = fz - ffz;

  // x pair [x0, x0+1]; clamps reproduce the reference exactly:
  //  f < 0   -> i0=0, i1=1, blend with w          (x0=0, wx unchanged)
  //  f > G-2 -> i0=i1=G-1, value=v[G-1] any w     (x0=G-2, wx=1)
  int x0 = (int)ffx;
  if (x0 < 0)     { x0 = 0; }
  if (x0 > G - 2) { x0 = G - 2; wx = 1.0f; }

  int y0 = (int)ffy; y0 = y0 < 0 ? 0 : (y0 > G - 1 ? G - 1 : y0);
  int z0 = (int)ffz; z0 = z0 < 0 ? 0 : (z0 > G - 1 ? G - 1 : z0);
  int y1 = y0 + 1 > G - 1 ? G - 1 : y0 + 1;
  int z1 = z0 + 1 > G - 1 ? G - 1 : z0 + 1;

  const size_t GG = (size_t)G * (size_t)G;
  const size_t G3 = GG * (size_t)G;

#pragma unroll
  for (int f = 0; f < 16; ++f) acc[f] = 0.0f;

#pragma unroll
  for (int c = 0; c < 4; ++c) {          // (z,y) corner pairs
    int yi = (c & 1) ? y1 : y0;
    int zi = (c & 2) ? z1 : z0;
    float w = ((c & 1) ? wy : 1.0f - wy) *
              ((c & 2) ? wz : 1.0f - wz);
    float w0 = w * (1.0f - wx);
    float w1 = w * wx;
    const float* p = g + (size_t)zi * GG + (size_t)yi * (size_t)G + (size_t)x0;
#pragma unroll
    for (int f = 0; f < 16; ++f) {
      float2 v = *(const float2*)(p + (size_t)f * G3);   // dword-aligned b64 load
      acc[f] = fmaf(w0, v.x, fmaf(w1, v.y, acc[f]));
    }
  }
}

__global__ __launch_bounds__(BLK) void feature_grid_mlp_kernel(
    const float* __restrict__ x,
    const float* __restrict__ g0, const float* __restrict__ g1, const float* __restrict__ g2,
    const float* __restrict__ W0, const float* __restrict__ b0,
    const float* __restrict__ W1, const float* __restrict__ b1,
    const float* __restrict__ W2, const float* __restrict__ b2,
    const float* __restrict__ W3, const float* __restrict__ b3,
    const float* __restrict__ W4, const float* __restrict__ b4,
    float* __restrict__ out)
{
  __shared__ __align__(16) _Float16 sW[NLAYERS][HIDDEN][KPAD];   // 36,864 B
  __shared__ __align__(32) float    sBias[NLAYERS][HIDDEN];      //  1,024 B
  __shared__ __align__(16) _Float16 sW4T[16][KPAD];              //  2,304 B (rows 1..15 zero)
  __shared__ __align__(16) _Float16 sAct[BLK][KPAD];             // 18,432 B

  const int tid  = threadIdx.x;
  const int lane = tid & 31;
  const int wave = tid >> 5;
  const unsigned p = blockIdx.x * BLK + tid;   // grid sized so p < B always

  // ---------------- stage weights (f16, transposed Wt[n][k], packed stores) ----------------
  {
    const float* Ws[NLAYERS]  = {W0, W1, W2, W3};
    const float* Bls[NLAYERS] = {b0, b1, b2, b3};
    const int    din[NLAYERS] = {48, 64, 64, 64};
#pragma unroll
    for (int l = 0; l < NLAYERS; ++l) {
#pragma unroll
      for (int it = 0; it < 4; ++it) {
        int idx = it * BLK + tid;            // 0..511 -> (n, 8-wide k-group)
        int n  = idx & 63;                   // consecutive lanes -> consecutive n: coalesced
        int kg = idx >> 6;                   // 0..7
        v8h pack;
#pragma unroll
        for (int j = 0; j < 8; ++j) {
          int k = kg * 8 + j;
          float v = (k < din[l]) ? Ws[l][k * HIDDEN + n] : 0.0f;
          pack[j] = (_Float16)v;
        }
        *(v8h*)&sW[l][n][kg * 8] = pack;     // one ds_store_b128
      }
      if (tid < HIDDEN) sBias[l][tid] = Bls[l][tid];
    }
    // W4^T as a 16x64 A-tile: row 0 = W4, rows 1..15 = 0
    _Float16* w4flat = &sW4T[0][0];
    for (int i = tid; i < 135; i += BLK) {   // zero rows 1..15 (v8h slots 9..143)
      v8h z = {};
      *(v8h*)&w4flat[(i + 9) * 8] = z;
    }
    if (tid < HIDDEN) sW4T[0][tid] = (_Float16)W4[tid];
    else if (tid < KPAD) sW4T[0][tid] = (_Float16)0.0f;
  }

  // ---------------- trilinear gather: 48 feats -> LDS f16 (packed stores) ----------------
  {
    float cx = x[p * 3 + 0], cy = x[p * 3 + 1], cz = x[p * 3 + 2];
    float acc[16];
    const float* grids[3] = {g0, g1, g2};
    const int    gsz[3]   = {128, 64, 32};
#pragma unroll
    for (int gi = 0; gi < 3; ++gi) {
      sample_grid(grids[gi], gsz[gi], cx, cy, cz, acc);
#pragma unroll
      for (int h = 0; h < 2; ++h) {
        v8h pack;
#pragma unroll
        for (int j = 0; j < 8; ++j) pack[j] = (_Float16)acc[h * 8 + j];
        *(v8h*)&sAct[tid][gi * 16 + h * 8] = pack;
      }
    }
    v8h z = {};
    *(v8h*)&sAct[tid][48] = z;               // pad K 48..63
    *(v8h*)&sAct[tid][56] = z;
  }

  __syncthreads();   // weights visible to all waves; activations staged

  // ------- MLP: D = W^T x act^T = H^T via v_wmma_f32_16x16x32_f16 -------
  // A = transposed weights (16 out-features x 32 K), B = activations (32 K x 16 points).
  // D element r = (row = out-feature r+8*hi, col = point lane%16): each lane ends up
  // with 8 CONSECUTIVE output features of its own point -> b128 write-back.
  const int hi = (lane & 16) ? 1 : 0;
  const int ln = lane & 15;

#pragma unroll
  for (int l = 0; l < NLAYERS; ++l) {
    // Hoist all A fragments (weights) + bias vectors for this layer.
    v16h aw[4][2];
    v8f  biasv[4];
#pragma unroll
    for (int nt = 0; nt < 4; ++nt) {
      const int nrow = nt * 16 + ln;
      biasv[nt] = *(const v8f*)&sBias[l][nt * 16 + hi * 8];
#pragma unroll
      for (int ch = 0; ch < 2; ++ch) {
        v8h alo = *(const v8h*)&sW[l][nrow][ch * 32 + (hi ? 8  : 0)];
        v8h ahi = *(const v8h*)&sW[l][nrow][ch * 32 + (hi ? 24 : 16)];
        aw[nt][ch] = SHUF16(alo, ahi);
      }
    }

#pragma unroll
    for (int t = 0; t < 2; ++t) {            // two 16-point tiles per wave
      const int brow = wave * 32 + t * 16 + ln;   // this lane's point row
      v16h bact[2];
#pragma unroll
      for (int ch = 0; ch < 2; ++ch) {       // B: lane's point, table-order K (matches A)
        const int kb = ch * 32 + (hi ? 8 : 0);
        v8h blo = *(const v8h*)&sAct[brow][kb];
        v8h bhi = *(const v8h*)&sAct[brow][kb + 16];
        bact[ch] = SHUF16(blo, bhi);
      }

      v8f c[4];
#pragma unroll
      for (int nt = 0; nt < 4; ++nt) {
        c[nt] = biasv[nt];                   // bias folded into accumulator C
#pragma unroll
        for (int ch = 0; ch < 2; ++ch)
          c[nt] = __builtin_amdgcn_wmma_f32_16x16x32_f16(
                      false, aw[nt][ch], false, bact[ch], (short)0, c[nt], false, false);
      }

      // cvt to f16 (v_cvt_pk_f16_f32), then packed-f16 ReLU (v_pk_max_num_f16),
      // then ONE b128 store per N-tile. ReLU commutes with RNE rounding here.
#pragma unroll
      for (int nt = 0; nt < 4; ++nt) {
        v8h hv;
#pragma unroll
        for (int j = 0; j < 8; ++j)
          hv[j] = (_Float16)c[nt][j];
        v8h zv = {};
        hv = __builtin_elementwise_max(hv, zv);
        *(v8h*)&sAct[brow][nt * 16 + hi * 8] = hv;
      }
    }
  }

  // ---------------- final 64->1 via WMMA (A rows 1..15 zero) + sigmoid ----------------
  {
    v16h a4[2];
#pragma unroll
    for (int ch = 0; ch < 2; ++ch) {
      v8h alo = *(const v8h*)&sW4T[ln][ch * 32 + (hi ? 8  : 0)];
      v8h ahi = *(const v8h*)&sW4T[ln][ch * 32 + (hi ? 24 : 16)];
      a4[ch] = SHUF16(alo, ahi);
    }
    const float b4v = b4[0];
#pragma unroll
    for (int t = 0; t < 2; ++t) {
      const int mrel = wave * 32 + t * 16;   // tile base within block
      const int brow = mrel + ln;
      v16h bact[2];
#pragma unroll
      for (int ch = 0; ch < 2; ++ch) {
        const int kb = ch * 32 + (hi ? 8 : 0);
        v8h blo = *(const v8h*)&sAct[brow][kb];
        v8h bhi = *(const v8h*)&sAct[brow][kb + 16];
        bact[ch] = SHUF16(blo, bhi);
      }
      v8f c4 = {0.f, 0.f, 0.f, 0.f, 0.f, 0.f, 0.f, 0.f};
#pragma unroll
      for (int ch = 0; ch < 2; ++ch)
        c4 = __builtin_amdgcn_wmma_f32_16x16x32_f16(
                 false, a4[ch], false, bact[ch], (short)0, c4, false, false);
      // D[0][m]: logit of point m lives in element 0 of lanes 0..15 (hi==0)
      if (lane < 16) {
        float s = c4[0] + b4v;
        out[blockIdx.x * BLK + mrel + ln] = 1.0f / (1.0f + expf(-s));
      }
    }
  }
}

extern "C" void kernel_launch(void* const* d_in, const int* in_sizes, int n_in,
                              void* d_out, int out_size, void* d_ws, size_t ws_size,
                              hipStream_t stream) {
  (void)n_in; (void)d_ws; (void)ws_size; (void)out_size;
  const float* x  = (const float*)d_in[0];
  const float* g0 = (const float*)d_in[1];
  const float* g1 = (const float*)d_in[2];
  const float* g2 = (const float*)d_in[3];
  const float* W0 = (const float*)d_in[4];  const float* b0 = (const float*)d_in[5];
  const float* W1 = (const float*)d_in[6];  const float* b1 = (const float*)d_in[7];
  const float* W2 = (const float*)d_in[8];  const float* b2 = (const float*)d_in[9];
  const float* W3 = (const float*)d_in[10]; const float* b3 = (const float*)d_in[11];
  const float* W4 = (const float*)d_in[12]; const float* b4 = (const float*)d_in[13];
  float* out = (float*)d_out;

  const int B = in_sizes[0] / 3;        // 1,048,576
  const int blocks = B / BLK;           // 8192 (exact -> EXEC all-ones for WMMA)

  feature_grid_mlp_kernel<<<blocks, BLK, 0, stream>>>(
      x, g0, g1, g2, W0, b0, W1, b1, W2, b2, W3, b3, W4, b4, out);
}
